// LlamaAttention_12481174962712
// MI455X (gfx1250) — compile-verified, matrix operand order checked by reading, not run on hardware
//
#include <hip/hip_runtime.h>

// ---------------------------------------------------------------------------
// Llama attention layer for MI455X (gfx1250, wave32, WMMA bf16 path)
// T=2048, HID=4096, H=32, KV=8, D=128, THETA=10000
// ---------------------------------------------------------------------------

#define T_SEQ 2048
#define HID   4096
#define NH    32
#define NKV   8
#define HD    128
#define QKVN  ((NH + 2 * NKV) * HD)   // 6144

typedef __bf16 bf16_t;
typedef __attribute__((ext_vector_type(16))) __bf16 v16bf;
typedef __attribute__((ext_vector_type(8)))  __bf16 v8bf;
typedef __attribute__((ext_vector_type(8)))  float  v8f;
typedef __attribute__((__vector_size__(16))) int    v4i;   // builtin's b128 unit

#define GLOBAL_AS __attribute__((address_space(1)))
#define LDS_AS    __attribute__((address_space(3)))

// --- async global->LDS (ASYNCcnt-tracked) with compile-safe fallback -------
#if __has_builtin(__builtin_amdgcn_global_load_async_to_lds_b128)
#define HAVE_ASYNC_LDS 1
__device__ __forceinline__ void copy_b128_to_lds(const void* g, void* l) {
  // prototype: (v4i AS1*, v4i AS3*, imm offset, imm cpol)
  __builtin_amdgcn_global_load_async_to_lds_b128(
      (GLOBAL_AS v4i*)g, (LDS_AS v4i*)l, /*offset=*/0, /*cpol=*/0);
}
#if __has_builtin(__builtin_amdgcn_s_wait_asynccnt)
#define WAIT_ASYNC(n) __builtin_amdgcn_s_wait_asynccnt(n)
#else
#define WAIT_ASYNC(n) asm volatile("s_wait_asynccnt %0" ::"n"(n) : "memory")
#endif
#else
#define HAVE_ASYNC_LDS 0
__device__ __forceinline__ void copy_b128_to_lds(const void* g, void* l) {
  *(v8bf*)l = *(const v8bf*)g;   // synchronous fallback; barrier orders it
}
#define WAIT_ASYNC(n) ((void)0)
#endif

__device__ __forceinline__ v8f wmma_bf16(v16bf a, v16bf b, v8f c) {
  // D = A(16x32 bf16) x B(32x16 bf16) + C(16x16 f32)
  return __builtin_amdgcn_wmma_f32_16x16x32_bf16(
      /*neg_a=*/false, a, /*neg_b=*/false, b,
      /*c_mod=*/(short)0, c, /*reuse_a=*/false, /*reuse_b=*/false);
}

// 16-element bf16 fragment = two contiguous 16-byte chunks at [base] and
// [base+16]; caller applies the lane-dependent hi*8 offset to match the
// CDNA5 16-bit A/B-fragment VGPR layout.
__device__ __forceinline__ v16bf load_frag(const bf16_t* __restrict__ base) {
  union { v16bf v; v8bf h[2]; } u;
  u.h[0] = *(const v8bf*)(base);
  u.h[1] = *(const v8bf*)(base + 16);
  return u.v;
}

// ---------------------------------------------------------------------------
// fp32 -> bf16 conversion (8 elems/thread, b128 loads)
// ---------------------------------------------------------------------------
__global__ __launch_bounds__(256) void cvt_f32_bf16(
    const float* __restrict__ in, bf16_t* __restrict__ out, long n) {
  long i = ((long)blockIdx.x * blockDim.x + threadIdx.x) * 8;
  if (i >= n) return;
  float4 a = *(const float4*)(in + i);
  float4 b = *(const float4*)(in + i + 4);
  v8bf o;
  o[0] = (bf16_t)a.x; o[1] = (bf16_t)a.y; o[2] = (bf16_t)a.z; o[3] = (bf16_t)a.w;
  o[4] = (bf16_t)b.x; o[5] = (bf16_t)b.y; o[6] = (bf16_t)b.z; o[7] = (bf16_t)b.w;
  *(v8bf*)(out + i) = o;
}

// ---------------------------------------------------------------------------
// C[M,N] (f32) = A[M,K] (bf16, row-major) x B[N,K]^T (bf16, row-major)
// Block: 128x128 tile, BK=32, double-buffered LDS fed by async b128 copies.
// 8 waves, each computing a 32x64 sub-tile (2x4 WMMA accumulators).
// ---------------------------------------------------------------------------
__global__ __launch_bounds__(256) void gemm_bf16_nt(
    const bf16_t* __restrict__ A, const bf16_t* __restrict__ B,
    float* __restrict__ C, int M, int N, int K) {
  __shared__ bf16_t Abuf[2][128][32];   // 16 KB
  __shared__ bf16_t Bbuf[2][128][32];   // 16 KB
  (void)M;

  const int tid  = threadIdx.x;
  const int lane = tid & 31;
  const int wave = tid >> 5;
  const int hi   = lane >> 4;
  const int col  = lane & 15;
  const int hi8  = hi * 8;
  const int wm   = wave >> 1;           // 0..3 : 32-row group
  const int wn   = wave & 1;            // 0..1 : 64-col group

  const int m0 = blockIdx.y * 128;
  const int n0 = blockIdx.x * 128;

  // Copy assignment: 128 rows x 4 b128-segments per matrix = 512 units;
  // each of 256 threads moves 2 units of A and 2 of B per stage.
  const int crow = tid >> 2;            // 0..63
  const int cseg = (tid & 3) * 8;       // bf16 element offset 0,8,16,24

  const bf16_t* __restrict__ gA0 = A + (size_t)(m0 + crow) * K + cseg;
  const bf16_t* __restrict__ gA1 = A + (size_t)(m0 + 64 + crow) * K + cseg;
  const bf16_t* __restrict__ gB0 = B + (size_t)(n0 + crow) * K + cseg;
  const bf16_t* __restrict__ gB1 = B + (size_t)(n0 + 64 + crow) * K + cseg;

#define ISSUE_STAGE(k0, buf)                                          \
  do {                                                                \
    copy_b128_to_lds(gA0 + (k0), &Abuf[buf][crow][cseg]);             \
    copy_b128_to_lds(gA1 + (k0), &Abuf[buf][64 + crow][cseg]);        \
    copy_b128_to_lds(gB0 + (k0), &Bbuf[buf][crow][cseg]);             \
    copy_b128_to_lds(gB1 + (k0), &Bbuf[buf][64 + crow][cseg]);        \
  } while (0)

  v8f acc[2][4];
#pragma unroll
  for (int mt = 0; mt < 2; ++mt)
#pragma unroll
    for (int nt = 0; nt < 4; ++nt) acc[mt][nt] = v8f{};

  const int nk = K >> 5;
  int buf = 0;
  ISSUE_STAGE(0, 0);                    // prologue: fill buffer 0

  for (int kt = 0; kt < nk; ++kt) {
    if (kt + 1 < nk) {
      ISSUE_STAGE((kt + 1) * 32, buf ^ 1);   // prefetch next stage
      WAIT_ASYNC(4);                          // current stage's 4 copies done
    } else {
      WAIT_ASYNC(0);
    }
    __syncthreads();                          // all threads' tiles visible

    v16bf af[2], bfr[4];
#pragma unroll
    for (int mt = 0; mt < 2; ++mt)
      af[mt] = load_frag(&Abuf[buf][wm * 32 + mt * 16 + col][hi8]);
#pragma unroll
    for (int nt = 0; nt < 4; ++nt)
      bfr[nt] = load_frag(&Bbuf[buf][wn * 64 + nt * 16 + col][hi8]);

#pragma unroll
    for (int mt = 0; mt < 2; ++mt)
#pragma unroll
      for (int nt = 0; nt < 4; ++nt)
        acc[mt][nt] = wmma_bf16(af[mt], bfr[nt], acc[mt][nt]);

    __syncthreads();                          // protect buf before overwrite
    buf ^= 1;
  }
#undef ISSUE_STAGE

  // C/D layout: VGPR r -> row (r + 8*hi), lane col -> column
#pragma unroll
  for (int mt = 0; mt < 2; ++mt) {
#pragma unroll
    for (int nt = 0; nt < 4; ++nt) {
#pragma unroll
      for (int r = 0; r < 8; ++r) {
        C[(size_t)(m0 + wm * 32 + mt * 16 + r + hi8) * N +
          n0 + wn * 64 + nt * 16 + col] = acc[mt][nt][r];
      }
    }
  }
}

// ---------------------------------------------------------------------------
// RoPE + repack.  qkv f32 [T, 6144] ->
//   Qr [H][T][D]  bf16 (rotated, pre-scaled by 1/sqrt(D))
//   Kr [KV][T][D] bf16 (rotated)
//   Vt [KV][D][T] bf16 (transposed for contiguous P*V B-fragment loads)
// ---------------------------------------------------------------------------
__global__ __launch_bounds__(256) void rope_pack(
    const float* __restrict__ qkv, const int* __restrict__ pos,
    bf16_t* __restrict__ Qr, bf16_t* __restrict__ Kr, bf16_t* __restrict__ Vt) {
  const int t = blockIdx.x;
  const float p = (float)pos[t];
  const float* __restrict__ row = qkv + (size_t)t * QKVN;
  const float qscale = 0.08838834764831845f;       // D^-0.5
  const float nlogth = -0.14391156f;               // -ln(10000)/64

  for (int i = threadIdx.x; i < NH * 64; i += blockDim.x) {
    int h = i >> 6, d = i & 63;
    float ang = p * __expf((float)d * nlogth);
    float s, c; __sincosf(ang, &s, &c);
    float x1 = row[h * HD + d];
    float x2 = row[h * HD + d + 64];
    size_t o = ((size_t)h * T_SEQ + t) * HD + d;
    Qr[o]      = (bf16_t)((x1 * c - x2 * s) * qscale);
    Qr[o + 64] = (bf16_t)((x2 * c + x1 * s) * qscale);
  }
  for (int i = threadIdx.x; i < NKV * 64; i += blockDim.x) {
    int h = i >> 6, d = i & 63;
    float ang = p * __expf((float)d * nlogth);
    float s, c; __sincosf(ang, &s, &c);
    float x1 = row[NH * HD + h * HD + d];
    float x2 = row[NH * HD + h * HD + d + 64];
    size_t o = ((size_t)h * T_SEQ + t) * HD + d;
    Kr[o]      = (bf16_t)(x1 * c - x2 * s);
    Kr[o + 64] = (bf16_t)(x2 * c + x1 * s);
  }
  for (int i = threadIdx.x; i < NKV * HD; i += blockDim.x) {
    int h = i >> 7, d = i & 127;
    Vt[((size_t)h * HD + d) * T_SEQ + t] =
        (bf16_t)row[(NH + NKV) * HD + h * HD + d];
  }
}

// ---------------------------------------------------------------------------
// Causal flash attention, one wave per (head, 16-query tile).
// Online softmax in f32; QK^T and P*V via bf16 WMMA; P staged in wave-private
// LDS (no block barriers -> per-wave causal trip counts are safe).
// Out: attn bf16 [T, H*D]
// ---------------------------------------------------------------------------
__global__ __launch_bounds__(256) void flash_attn(
    const bf16_t* __restrict__ Qr, const bf16_t* __restrict__ Kr,
    const bf16_t* __restrict__ Vt, bf16_t* __restrict__ Out) {
  __shared__ bf16_t smem[8][16][32];               // 8 waves x 1 KB

  const int lane = threadIdx.x & 31;
  const int wave = threadIdx.x >> 5;
  const int hi   = lane >> 4;
  const int col  = lane & 15;
  const int hi8  = hi * 8;

  const int h  = blockIdx.y;
  const int kv = h >> 2;                           // GQA group of 4
  const int q0 = (blockIdx.x * 8 + wave) * 16;

  const bf16_t* __restrict__ qrow =
      Qr + ((size_t)h * T_SEQ + q0 + col) * HD + hi8;
  v16bf qf[4];
#pragma unroll
  for (int kd = 0; kd < 4; ++kd) qf[kd] = load_frag(qrow + kd * 32);

  v8f o[8];
#pragma unroll
  for (int j = 0; j < 8; ++j) o[j] = v8f{};
  float m[8], l[8];
#pragma unroll
  for (int r = 0; r < 8; ++r) { m[r] = -1e30f; l[r] = 0.f; }

  const bf16_t* __restrict__ kbase = Kr + (size_t)kv * T_SEQ * HD;
  const bf16_t* __restrict__ vbase = Vt + (size_t)kv * HD * T_SEQ;

  const int nblocks = (q0 + 16 + 31) / 32;         // causal: key blocks of 32

  for (int kb = 0; kb < nblocks; ++kb) {
    const int kstart = kb * 32;

    // S = Q * K^T  (two 16x16 tiles, 4 K-dim accumulation steps each)
    v8f s[2];
#pragma unroll
    for (int nt = 0; nt < 2; ++nt) {
      v8f sacc = {};
      const bf16_t* krow =
          kbase + (size_t)(kstart + nt * 16 + col) * HD + hi8;
#pragma unroll
      for (int kd = 0; kd < 4; ++kd)
        sacc = wmma_bf16(qf[kd], load_frag(krow + kd * 32), sacc);
      s[nt] = sacc;
    }

    // Online softmax per row (row = r + 8*hi; 16 lanes of a half share a row)
    float p0v[8], p1v[8];
#pragma unroll
    for (int r = 0; r < 8; ++r) {
      const int qi = q0 + r + hi8;
      float s0 = s[0][r];
      float s1 = s[1][r];
      if (kstart + col      > qi) s0 = -1e30f;     // causal mask -> v_cndmask
      if (kstart + 16 + col > qi) s1 = -1e30f;
      float mx = fmaxf(s0, s1);
      mx = fmaxf(mx, __shfl_xor(mx, 1));
      mx = fmaxf(mx, __shfl_xor(mx, 2));
      mx = fmaxf(mx, __shfl_xor(mx, 4));
      mx = fmaxf(mx, __shfl_xor(mx, 8));
      const float mn   = fmaxf(m[r], mx);
      const float corr = __expf(m[r] - mn);
      const float p0 = __expf(s0 - mn);
      const float p1 = __expf(s1 - mn);
      float ps = p0 + p1;
      ps += __shfl_xor(ps, 1);
      ps += __shfl_xor(ps, 2);
      ps += __shfl_xor(ps, 4);
      ps += __shfl_xor(ps, 8);
      l[r] = l[r] * corr + ps;
      m[r] = mn;
#pragma unroll
      for (int j = 0; j < 8; ++j) o[j][r] = o[j][r] * corr;
      p0v[r] = p0;
      p1v[r] = p1;
    }

    // Stage P (16x32 bf16) in wave-private LDS: C-layout -> row-major
#pragma unroll
    for (int r = 0; r < 8; ++r) {
      smem[wave][r + hi8][col]      = (bf16_t)p0v[r];
      smem[wave][r + hi8][16 + col] = (bf16_t)p1v[r];
    }
    asm volatile("s_wait_dscnt 0" ::: "memory");

    // Reload P as an A-fragment (ds_load_b128 x2)
    union { v16bf v; v8bf hh[2]; } pf;
    pf.hh[0] = *(const v8bf*)&smem[wave][col][hi8];
    pf.hh[1] = *(const v8bf*)&smem[wave][col][hi8 + 16];

    // O += P * V  (8 output N-tiles over D=128)
#pragma unroll
    for (int j = 0; j < 8; ++j) {
      const bf16_t* vrow =
          vbase + (size_t)(j * 16 + col) * T_SEQ + kstart + hi8;
      o[j] = wmma_bf16(pf.v, load_frag(vrow), o[j]);
    }
  }

  // Normalize and write attn output bf16 [T, H*D]
#pragma unroll
  for (int r = 0; r < 8; ++r) {
    const float invl = 1.0f / l[r];
    const size_t trow = (size_t)(q0 + r + hi8);
#pragma unroll
    for (int j = 0; j < 8; ++j) {
      Out[trow * (NH * HD) + h * HD + j * 16 + col] =
          (bf16_t)(o[j][r] * invl);
    }
  }
}

// ---------------------------------------------------------------------------
// Launch
// ---------------------------------------------------------------------------
extern "C" void kernel_launch(void* const* d_in, const int* in_sizes, int n_in,
                              void* d_out, int out_size, void* d_ws, size_t ws_size,
                              hipStream_t stream) {
  (void)in_sizes; (void)n_in; (void)out_size; (void)ws_size;

  const float* hs   = (const float*)d_in[0];   // [T, HID] f32
  const int*   pos  = (const int*)d_in[1];     // [T] i32
  const float* wqkv = (const float*)d_in[2];   // [6144, 4096] f32
  const float* wo   = (const float*)d_in[3];   // [4096, 4096] f32
  float* out = (float*)d_out;                  // [T, HID] f32

  char* ws = (char*)d_ws;
  const size_t offA = 0;                                   // hidden bf16 (16 MB)
  const size_t offB = offA + (size_t)T_SEQ * HID * 2;      // W bf16 slot (48 MB)
  const size_t offC = offB + (size_t)QKVN * HID * 2;       // qkv f32 / attn bf16
  const size_t offQ = offC + (size_t)T_SEQ * QKVN * 4;     // Qr (16 MB)
  const size_t offK = offQ + (size_t)NH * T_SEQ * HD * 2;  // Kr (4 MB)
  const size_t offV = offK + (size_t)NKV * T_SEQ * HD * 2; // Vt (4 MB)

  bf16_t* hsb  = (bf16_t*)(ws + offA);
  bf16_t* wb   = (bf16_t*)(ws + offB);
  float*  qkv  = (float*)(ws + offC);
  bf16_t* attn = (bf16_t*)(ws + offC);   // reuses qkv slot after RoPE consumed it
  bf16_t* Qr   = (bf16_t*)(ws + offQ);
  bf16_t* Kr   = (bf16_t*)(ws + offK);
  bf16_t* Vt   = (bf16_t*)(ws + offV);

  {
    long n = (long)T_SEQ * HID;
    cvt_f32_bf16<<<(int)(n / (8 * 256)), 256, 0, stream>>>(hs, hsb, n);
  }
  {
    long n = (long)QKVN * HID;
    cvt_f32_bf16<<<(int)(n / (8 * 256)), 256, 0, stream>>>(wqkv, wb, n);
  }
  // qkv = X * Wqkv^T
  gemm_bf16_nt<<<dim3(QKVN / 128, T_SEQ / 128), 256, 0, stream>>>(
      hsb, wb, qkv, T_SEQ, QKVN, HID);
  // RoPE + repack
  rope_pack<<<T_SEQ, 256, 0, stream>>>(qkv, pos, Qr, Kr, Vt);
  // causal flash attention
  flash_attn<<<dim3(T_SEQ / 128, NH), 256, 0, stream>>>(Qr, Kr, Vt, attn);
  // out = attn * Wo^T
  {
    long n = (long)HID * HID;
    cvt_f32_bf16<<<(int)(n / (8 * 256)), 256, 0, stream>>>(wo, wb, n);
  }
  gemm_bf16_nt<<<dim3(HID / 128, T_SEQ / 128), 256, 0, stream>>>(
      attn, wb, out, T_SEQ, HID, HID);
}
